// aoGPT_28887950033546
// MI455X (gfx1250) — matrix operand order repaired and checked by reading, not compiled, not run on hardware
//
#include <hip/hip_runtime.h>
#include <hip/hip_bf16.h>
#include <math.h>

// ---------------------------------------------------------------------------
// aoGPT forward for MI455X (gfx1250, wave32, WMMA + TDM).
// All dense math via v_wmma_f32_16x16x32_bf16 (bf16 in, f32 acc).
// GEMM A tiles staged into LDS by the Tensor Data Mover (tensor_load_to_lds,
// TENSORcnt-tracked, double buffered); fallback is vectorized b128 copies.
// V is stored transposed [B,H,E,S] so attention PV fragments are contiguous.
// ---------------------------------------------------------------------------

typedef __bf16 bf16_t;
typedef __attribute__((ext_vector_type(16))) __bf16 v16bf;
typedef __attribute__((ext_vector_type(8)))  __bf16 v8bf;
typedef __attribute__((ext_vector_type(8)))  float  v8f;

static constexpr int kNL  = 6;
static constexpr int kB   = 2;
static constexpr int kT   = 1024;
static constexpr int kLC  = 128;
static constexpr int kS   = 1152;   // LC + 1 + (T-1)
static constexpr int kC   = 768;
static constexpr int kH   = 12;
static constexpr int kE   = 64;
static constexpr int kV   = 32002;
static constexpr int kM   = kB * kS;   // 2304 GEMM rows
static constexpr int kPAD = 32000;
static constexpr int kSEP = 32001;

// ---------------------------------------------------------------------------
// TDM (Tensor Data Mover) support — 6-arg builtin (clang-23 signature):
// (uint32x4 g0, int32x8 g1, int32x4 g2, int32x4 g3, int32x8 g4, i32 cpol)
// ---------------------------------------------------------------------------
#if __has_builtin(__builtin_amdgcn_tensor_load_to_lds) && \
    __has_builtin(__builtin_amdgcn_s_wait_tensorcnt)
#define USE_TDM 1
typedef unsigned __attribute__((ext_vector_type(4))) tdm_u4;
typedef int      __attribute__((ext_vector_type(8))) tdm_i8;
typedef int      __attribute__((ext_vector_type(4))) tdm_i4;

// Issue a 2-D TDM tile load: tile (tile_rows x 64 el, bf16) from a row-major
// tensor with row stride `stride_el`, into LDS at byte offset `lds_off`.
// D# bit packing per CDNA5 ISA 8.3/8.4.
__device__ __forceinline__ void tdm_load_tile(const bf16_t* gsrc, unsigned lds_off,
                                              int rows_avail, int cols_avail,
                                              int tile_rows, int stride_el) {
  const unsigned long long ga = (unsigned long long)(size_t)gsrc;
  tdm_u4 g0;
  g0[0] = 1u;                                             // count=1, user mode
  g0[1] = lds_off;                                        // lds_addr (bytes)
  g0[2] = (unsigned)(ga & 0xffffffffull);                 // global_addr lo
  g0[3] = (unsigned)((ga >> 32) & 0x01fffffful)           // global_addr hi (56:32)
        | (2u << 30);                                     // type=2 ("image")
  const unsigned td0 = (unsigned)cols_avail;              // tensor_dim0 (OOB bound)
  const unsigned td1 = (unsigned)rows_avail;              // tensor_dim1
  const unsigned st0 = (unsigned)stride_el;               // tensor_dim0_stride
  tdm_i8 g1;
  g1[0] = (int)(1u << 16);                                // data_size=1 (2 bytes)
  g1[1] = (int)((td0 & 0xffffu) << 16);                   // td0[15:0] -> bits 63:48
  g1[2] = (int)((td0 >> 16) | ((td1 & 0xffffu) << 16));   // td0[31:16], td1[15:0]
  g1[3] = (int)((td1 >> 16) | (64u << 16));               // td1[31:16], tile_dim0=64
  g1[4] = (int)((unsigned)tile_rows & 0xffffu);           // tile_dim1, tile_dim2=0
  g1[5] = (int)st0;                                       // dim0 stride lo32
  g1[6] = 0;                                              // dim0 stride hi16
  g1[7] = 0;                                              // dim1 stride hi32
  tdm_i4 z4 = {};                                         // groups 2/3: 2-D tensor
  tdm_i8 z8 = {};                                         // group 4 (clang-23 form)
  __builtin_amdgcn_tensor_load_to_lds(g0, g1, z4, z4, z8, 0);
}
#endif

// ---------------------------------------------------------------------------
// WMMA + fragment helpers (layouts per CDNA5 ISA 7.12.2)
// ---------------------------------------------------------------------------
__device__ __forceinline__ v8f wmma_bf16(v16bf a, v16bf b, v8f c) {
  return __builtin_amdgcn_wmma_f32_16x16x32_bf16(false, a, false, b,
                                                 (short)0, c, false, false);
}

__device__ __forceinline__ v16bf cat8(v8bf lo, v8bf hi) {
  return __builtin_shufflevector(lo, hi, 0, 1, 2, 3, 4, 5, 6, 7,
                                 8, 9, 10, 11, 12, 13, 14, 15);
}

// A-matrix 16x32 (MxK) from row-major source (global or LDS).
__device__ __forceinline__ v16bf load_fragA(const bf16_t* __restrict__ A,
                                            int lda, int row0, int k0, int lane) {
  const int g = lane >> 4, mm = lane & 15;
  const bf16_t* Ar = A + (size_t)(row0 + mm) * lda + k0 + g * 8;
  return cat8(*(const v8bf*)Ar, *(const v8bf*)(Ar + 16));
}

// B-matrix 32x16 (KxN) with B[k][n] = W[n][k], W row-major [N,K].
__device__ __forceinline__ v16bf load_fragB_wt(const bf16_t* __restrict__ W,
                                               int ldw, int n0, int k0,
                                               int lane, int N) {
  const int g = lane >> 4, nn = n0 + (lane & 15);
  if (nn < N) {
    const bf16_t* Wr = W + (size_t)nn * ldw + k0 + g * 8;
    return cat8(*(const v8bf*)Wr, *(const v8bf*)(Wr + 16));
  }
  v8bf z = {};
  return cat8(z, z);
}

// ---------------------------------------------------------------------------
// Input packing
// ---------------------------------------------------------------------------
__global__ void build_inputs_kernel(const int* __restrict__ idx,
                                    const int* __restrict__ order,
                                    int* __restrict__ iidx,
                                    int* __restrict__ ipos) {
  const int b = blockIdx.x;
  if (threadIdx.x != 0) return;
  const int* ord = order + (size_t)b * kT;
  int blen = 0;
  for (int t = 0; t + 1 < kT; ++t) {
    int d = ord[t + 1] - ord[t];
    if (d < 0) d = -d;
    if (d != 1) { blen = t + 1; break; }
  }
  const int off = kLC - blen;
  for (int j = 0; j < kLC; ++j) {
    const bool valid = j >= off;
    int gi = j - off;
    gi = gi < 0 ? 0 : (gi > kT - 1 ? kT - 1 : gi);
    const int ci = ord[gi];
    iidx[b * kS + j] = valid ? idx[b * kT + ci] : kPAD;
    ipos[b * kS + j] = valid ? ci : 0;
  }
  iidx[b * kS + kLC] = kSEP;
  ipos[b * kS + kLC] = 0;
  for (int t = 0; t < kT - 1; ++t) {
    iidx[b * kS + kLC + 1 + t] = idx[b * kT + t];
    ipos[b * kS + kLC + 1 + t] = t;
  }
}

// ---------------------------------------------------------------------------
// Embedding + sinusoidal positional encoding
// ---------------------------------------------------------------------------
__global__ void embed_kernel(const int* __restrict__ iidx,
                             const int* __restrict__ ipos,
                             const float* __restrict__ wte,
                             float* __restrict__ x) {
  const int row = blockIdx.x;
  const int tok = iidx[row];
  const float fpos = (float)ipos[row];
  const float* wr = wte + (size_t)tok * kC;
  for (int c = threadIdx.x; c < kC; c += blockDim.x) {
    const int i2 = (c >> 1) << 1;
    const float d = __expf(-(float)i2 * (9.210340371976184f / (float)kC));
    const float pe = (c & 1) ? __cosf(fpos * d) : __sinf(fpos * d);
    x[(size_t)row * kC + c] = wr[c] + pe;
  }
}

__global__ void cvt_f32_bf16_kernel(const float* __restrict__ src,
                                    bf16_t* __restrict__ dst, size_t n) {
  for (size_t i = blockIdx.x * (size_t)blockDim.x + threadIdx.x; i < n;
       i += (size_t)gridDim.x * blockDim.x)
    dst[i] = (bf16_t)src[i];
}

// ---------------------------------------------------------------------------
// LayerNorm (f32 in) -> bf16 out
// ---------------------------------------------------------------------------
__global__ void layernorm_kernel(const float* __restrict__ x,
                                 const float* __restrict__ w,
                                 const float* __restrict__ bvec,
                                 bf16_t* __restrict__ out) {
  const int row = blockIdx.x;
  const float* xr = x + (size_t)row * kC;
  __shared__ float sred[256];
  float s = 0.f;
  for (int c = threadIdx.x; c < kC; c += blockDim.x) s += xr[c];
  sred[threadIdx.x] = s;
  __syncthreads();
  for (int off = blockDim.x >> 1; off; off >>= 1) {
    if ((int)threadIdx.x < off) sred[threadIdx.x] += sred[threadIdx.x + off];
    __syncthreads();
  }
  const float mean = sred[0] / (float)kC;
  __syncthreads();
  float v = 0.f;
  for (int c = threadIdx.x; c < kC; c += blockDim.x) {
    const float d = xr[c] - mean;
    v += d * d;
  }
  sred[threadIdx.x] = v;
  __syncthreads();
  for (int off = blockDim.x >> 1; off; off >>= 1) {
    if ((int)threadIdx.x < off) sred[threadIdx.x] += sred[threadIdx.x + off];
    __syncthreads();
  }
  const float rstd = rsqrtf(sred[0] / (float)kC + 1e-5f);
  for (int c = threadIdx.x; c < kC; c += blockDim.x)
    out[(size_t)row * kC + c] = (bf16_t)((xr[c] - mean) * rstd * w[c] + bvec[c]);
}

// ---------------------------------------------------------------------------
// WMMA GEMM: out[M,N] = A[M,K] @ W[N,K]^T (+ bias), fused epilogues.
// 128 threads = 4 waves; block tile 32(M) x 64(N); A double-buffered in LDS
// via TDM (wave 0 issues, s_wait_tensorcnt, barrier) or b128 copy fallback.
// ---------------------------------------------------------------------------
template <int MODE>
__global__ void gemm16_kernel(const bf16_t* __restrict__ A,
                              const bf16_t* __restrict__ W,
                              const float* __restrict__ bias,
                              void* out0, void* out1, void* out2,
                              int M, int N, int K) {
  const int lane = threadIdx.x & 31;
  const int wv   = threadIdx.x >> 5;                // wave 0..3
  const int row0 = blockIdx.y * 32;
  const int n0   = (blockIdx.x * 4 + wv) * 16;
  const int g = lane >> 4, nloc = lane & 15, n = n0 + nloc;

  __shared__ __align__(16) bf16_t sA[2][32][64];    // 8 KB, double buffered

#ifdef USE_TDM
  auto stage = [&](int buf, int kb) {
    if (wv == 0 && lane == 0)
      tdm_load_tile(A + (size_t)row0 * K + kb,
                    (unsigned)(size_t)&sA[buf][0][0],
                    M - row0, K - kb, 32, K);
  };
  auto stage_wait = [&](bool in_flight) {
    if (wv == 0 && lane == 0) {
      if (in_flight) __builtin_amdgcn_s_wait_tensorcnt(1);   // keep next in flight
      else           __builtin_amdgcn_s_wait_tensorcnt(0);
    }
  };
#else
  const int co   = (int)threadIdx.x * 16;           // 128 thr * 16 el = 32*64
  const int crow = co >> 6, ccol = co & 63;
  const bf16_t* Asrc = A + (size_t)(row0 + crow) * K + ccol;
  auto stage = [&](int buf, int kb) {
    const bf16_t* s = Asrc + kb;
    *(v8bf*)&sA[buf][crow][ccol]     = *(const v8bf*)s;
    *(v8bf*)&sA[buf][crow][ccol + 8] = *(const v8bf*)(s + 8);
  };
  auto stage_wait = [&](bool) {};
#endif

  v8f acc0 = {}, acc1 = {};
  stage(0, 0);
  for (int kb = 0; kb < K; kb += 64) {
    const int cur = (kb >> 6) & 1;
    const bool havenext = (kb + 64 < K);
    if (havenext) {
      stage(cur ^ 1, kb + 64);
      if (n < N) __builtin_prefetch(W + (size_t)n * K + kb + 64, 0, 1);
    }
    stage_wait(havenext);
    __syncthreads();                                 // tile visible to all waves
#pragma unroll
    for (int kk = 0; kk < 64; kk += 32) {
      const v16bf bf = load_fragB_wt(W, K, n0, kb + kk, lane, N);
      const v16bf a0 = load_fragA(&sA[cur][0][0], 64, 0,  kk, lane);
      const v16bf a1 = load_fragA(&sA[cur][0][0], 64, 16, kk, lane);
      acc0 = wmma_bf16(a0, bf, acc0);
      acc1 = wmma_bf16(a1, bf, acc1);
    }
    __syncthreads();                                 // done reading before reuse
  }
  if (n >= N) return;  // tail guard (logits N=32002) — after all WMMAs
  const float bv = bias ? bias[n] : 0.f;

#pragma unroll
  for (int rt = 0; rt < 2; ++rt) {
    const v8f acc = rt ? acc1 : acc0;
    const int rbase = row0 + rt * 16 + 8 * g;       // 8 consecutive rows
    if (MODE == 3) {
      const int part = n / kC, cc = n - part * kC, hh = cc >> 6, e = cc & 63;
      const int bb = rbase / kS, sbase = rbase - bb * kS;
      if (part == 2) {   // V transposed: one aligned 16B store per lane
        v8bf pack;
#pragma unroll
        for (int r = 0; r < 8; ++r) pack[r] = (bf16_t)(acc[r] + bv);
        *(v8bf*)((bf16_t*)out2 +
                 (((size_t)bb * kH + hh) * kE + e) * kS + sbase) = pack;
      } else {
        bf16_t* dst = part == 0 ? (bf16_t*)out0 : (bf16_t*)out1;
#pragma unroll
        for (int r = 0; r < 8; ++r)
          dst[(((size_t)bb * kH + hh) * kS + (sbase + r)) * kE + e] =
              (bf16_t)(acc[r] + bv);
      }
    } else {
#pragma unroll
      for (int r = 0; r < 8; ++r) {
        const int m = rbase + r;
        float val = acc[r] + bv;
        if (MODE == 0) {
          ((bf16_t*)out0)[(size_t)m * N + n] = (bf16_t)val;
        } else if (MODE == 1) {
          val = 0.5f * val * (1.f + erff(val * 0.7071067811865476f));
          ((bf16_t*)out0)[(size_t)m * N + n] = (bf16_t)val;
        } else if (MODE == 2) {
          ((float*)out0)[(size_t)m * N + n] += val;
        } else {
          ((float*)out0)[(size_t)m * N + n] = val;
        }
      }
    }
  }
}

// ---------------------------------------------------------------------------
// Flash-style causal attention (one wave per (b, head, 16-query tile)).
// ---------------------------------------------------------------------------
__global__ void attention_kernel(const bf16_t* __restrict__ q,
                                 const bf16_t* __restrict__ k,
                                 const bf16_t* __restrict__ vT,
                                 const int* __restrict__ iidx,
                                 bf16_t* __restrict__ y) {
  const int lane = threadIdx.x;
  const int q0 = blockIdx.x * 16, head = blockIdx.y, b = blockIdx.z;
  const int g = lane >> 4, nloc = lane & 15;
  const size_t hb = (((size_t)b * kH + head) * kS) * kE;
  const bf16_t* Qh  = q + hb;
  const bf16_t* Kh  = k + hb;
  const bf16_t* VTh = vT + (((size_t)b * kH + head) * kE) * kS;  // [E][S]
  const v16bf qA0 = load_fragA(Qh, kE, q0, 0, lane);
  const v16bf qA1 = load_fragA(Qh, kE, q0, 32, lane);
  __shared__ __align__(16) bf16_t Pt[16][32];
  v8f o0 = {}, o1 = {}, o2 = {}, o3 = {};
  float m8[8], l8[8];
#pragma unroll
  for (int r = 0; r < 8; ++r) { m8[r] = -1e30f; l8[r] = 0.f; }

  for (int k0 = 0; k0 < q0 + 16; k0 += 32) {
    v8f sa = {}, sb = {};
    sa = wmma_bf16(qA0, load_fragB_wt(Kh, kE, k0, 0, lane, kS), sa);
    sa = wmma_bf16(qA1, load_fragB_wt(Kh, kE, k0, 32, lane, kS), sa);
    sb = wmma_bf16(qA0, load_fragB_wt(Kh, kE, k0 + 16, 0, lane, kS), sb);
    sb = wmma_bf16(qA1, load_fragB_wt(Kh, kE, k0 + 16, 32, lane, kS), sb);
    const int key0 = k0 + nloc, key1 = k0 + 16 + nloc;
    const bool pad0 = iidx[b * kS + key0] == kPAD;
    const bool pad1 = iidx[b * kS + key1] == kPAD;
#pragma unroll
    for (int r = 0; r < 8; ++r) {
      const int qg = q0 + r + 8 * g;
      const bool c0 = key0 <= qg, c1 = key1 <= qg;
      const float s0 = c0 ? sa[r] * 0.125f : -1e30f;   // scale = 1/sqrt(64)
      const float s1 = c1 ? sb[r] * 0.125f : -1e30f;
      float ml = fmaxf(s0, s1);
#pragma unroll
      for (int off = 8; off; off >>= 1) ml = fmaxf(ml, __shfl_xor(ml, off, 32));
      const float mnew = fmaxf(m8[r], ml);
      const float alpha = __expf(m8[r] - mnew);
      m8[r] = mnew;
      const float p0 = c0 ? __expf(s0 - mnew) : 0.f;
      const float p1 = c1 ? __expf(s1 - mnew) : 0.f;
      float rs = p0 + p1;   // denominator keeps PAD keys (reference semantics)
#pragma unroll
      for (int off = 8; off; off >>= 1) rs += __shfl_xor(rs, off, 32);
      l8[r] = l8[r] * alpha + rs;
      o0[r] *= alpha; o1[r] *= alpha; o2[r] *= alpha; o3[r] *= alpha;
      Pt[r + 8 * g][nloc]      = (bf16_t)(pad0 ? 0.f : p0);  // numerator drops PAD
      Pt[r + 8 * g][16 + nloc] = (bf16_t)(pad1 ? 0.f : p1);
    }
    __syncthreads();
    const v16bf pA = load_fragA(&Pt[0][0], 32, 0, 0, lane);
    o0 = wmma_bf16(pA, load_fragB_wt(VTh, kS, 0,  k0, lane, kE), o0);
    o1 = wmma_bf16(pA, load_fragB_wt(VTh, kS, 16, k0, lane, kE), o1);
    o2 = wmma_bf16(pA, load_fragB_wt(VTh, kS, 32, k0, lane, kE), o2);
    o3 = wmma_bf16(pA, load_fragB_wt(VTh, kS, 48, k0, lane, kE), o3);
    __syncthreads();
  }
#pragma unroll
  for (int r = 0; r < 8; ++r) {
    const size_t m = (size_t)b * kS + q0 + r + 8 * g;
    const float inv = 1.f / l8[r];
    bf16_t* yr = y + m * kC + head * kE + nloc;
    yr[0]  = (bf16_t)(o0[r] * inv);
    yr[16] = (bf16_t)(o1[r] * inv);
    yr[32] = (bf16_t)(o2[r] * inv);
    yr[48] = (bf16_t)(o3[r] * inv);
  }
}

// ---------------------------------------------------------------------------
// Host orchestration
// ---------------------------------------------------------------------------
extern "C" void kernel_launch(void* const* d_in, const int* in_sizes, int n_in,
                              void* d_out, int out_size, void* d_ws, size_t ws_size,
                              hipStream_t stream) {
  const int*   idx    = (const int*)d_in[0];
  const int*   order  = (const int*)d_in[1];
  const float* wte    = (const float*)d_in[3];
  const float* ln1_w  = (const float*)d_in[4];
  const float* ln1_b  = (const float*)d_in[5];
  const float* attn_w = (const float*)d_in[6];
  const float* attn_b = (const float*)d_in[7];
  const float* proj_w = (const float*)d_in[8];
  const float* proj_b = (const float*)d_in[9];
  const float* ln2_w  = (const float*)d_in[10];
  const float* ln2_b  = (const float*)d_in[11];
  const float* fc_w   = (const float*)d_in[12];
  const float* fc_b   = (const float*)d_in[13];
  const float* fcp_w  = (const float*)d_in[14];
  const float* fcp_b  = (const float*)d_in[15];
  const float* lnf_w  = (const float*)d_in[16];
  const float* lnf_b  = (const float*)d_in[17];

  char* basep = (char*)d_ws;
  size_t off = 0;
  auto alloc = [&](size_t bytes) -> void* {
    off = (off + 255) & ~(size_t)255;
    void* p = basep + off;
    off += bytes;
    return p;
  };
  int*    iidx  = (int*)alloc((size_t)kM * 4);
  int*    ipos  = (int*)alloc((size_t)kM * 4);
  float*  x     = (float*)alloc((size_t)kM * kC * 4);
  bf16_t* h     = (bf16_t*)alloc((size_t)kM * kC * 2);
  bf16_t* y     = (bf16_t*)alloc((size_t)kM * kC * 2);
  bf16_t* qb    = (bf16_t*)alloc((size_t)kM * kC * 2);
  bf16_t* kb    = (bf16_t*)alloc((size_t)kM * kC * 2);
  bf16_t* vtb   = (bf16_t*)alloc((size_t)kM * kC * 2);   // V^T [B,H,E,S]
  bf16_t* mb    = (bf16_t*)alloc((size_t)kM * 4 * kC * 2);
  bf16_t* wAttn = (bf16_t*)alloc((size_t)3 * kC * kC * 2);
  bf16_t* wProj = (bf16_t*)alloc((size_t)kC * kC * 2);
  bf16_t* wFc   = (bf16_t*)alloc((size_t)4 * kC * kC * 2);
  bf16_t* wFcp  = (bf16_t*)alloc((size_t)kC * 4 * kC * 2);
  bf16_t* wteBf = (bf16_t*)alloc((size_t)kV * kC * 2);

  auto cvt = [&](const float* s, bf16_t* d, size_t n) {
    int blocks = (int)((n + 255) / 256);
    if (blocks > 16384) blocks = 16384;
    cvt_f32_bf16_kernel<<<blocks, 256, 0, stream>>>(s, d, n);
  };

  build_inputs_kernel<<<kB, 32, 0, stream>>>(idx, order, iidx, ipos);
  embed_kernel<<<kM, 256, 0, stream>>>(iidx, ipos, wte, x);
  cvt(wte, wteBf, (size_t)kV * kC);

  const dim3 blk(128);
  for (int l = 0; l < kNL; ++l) {
    cvt(attn_w + (size_t)l * 3 * kC * kC, wAttn, (size_t)3 * kC * kC);
    cvt(proj_w + (size_t)l * kC * kC,     wProj, (size_t)kC * kC);
    cvt(fc_w   + (size_t)l * 4 * kC * kC, wFc,   (size_t)4 * kC * kC);
    cvt(fcp_w  + (size_t)l * kC * 4 * kC, wFcp,  (size_t)kC * 4 * kC);

    layernorm_kernel<<<kM, 256, 0, stream>>>(x, ln1_w + l * kC, ln1_b + l * kC, h);
    gemm16_kernel<3><<<dim3(3 * kC / 64, kM / 32), blk, 0, stream>>>(
        h, wAttn, attn_b + (size_t)l * 3 * kC, qb, kb, vtb, kM, 3 * kC, kC);
    attention_kernel<<<dim3(kS / 16, kH, kB), dim3(32), 0, stream>>>(
        qb, kb, vtb, iidx, y);
    gemm16_kernel<2><<<dim3(kC / 64, kM / 32), blk, 0, stream>>>(
        y, wProj, proj_b + (size_t)l * kC, x, nullptr, nullptr, kM, kC, kC);

    layernorm_kernel<<<kM, 256, 0, stream>>>(x, ln2_w + l * kC, ln2_b + l * kC, h);
    gemm16_kernel<1><<<dim3(4 * kC / 64, kM / 32), blk, 0, stream>>>(
        h, wFc, fc_b + (size_t)l * 4 * kC, mb, nullptr, nullptr, kM, 4 * kC, kC);
    gemm16_kernel<2><<<dim3(kC / 64, kM / 32), blk, 0, stream>>>(
        mb, wFcp, fcp_b + (size_t)l * kC, x, nullptr, nullptr, kM, kC, 4 * kC);
  }

  layernorm_kernel<<<kM, 256, 0, stream>>>(x, lnf_w, lnf_b, h);
  gemm16_kernel<4><<<dim3((kV + 63) / 64, kM / 32), blk, 0, stream>>>(
      h, wteBf, nullptr, d_out, nullptr, nullptr, kM, kV, kC);
}
